// GNNEncoder_29008209117550
// MI455X (gfx1250) — matrix operand order repaired and checked by reading, not compiled, not run on hardware
//
#include <hip/hip_runtime.h>
#include <hip/hip_bf16.h>

typedef __attribute__((ext_vector_type(16))) _Float16 v16h;
typedef __attribute__((ext_vector_type(8)))  _Float16 v8h;
typedef __attribute__((ext_vector_type(4)))  _Float16 v4h;
typedef __attribute__((ext_vector_type(8)))  float    v8f;

#define LN_EPS 1e-5f

// ---------------------------------------------------------------------------
// 1) Input projection: h0 = relu(x @ W_in + b_in);  x:(N,6), W_in:(6,64)
// ---------------------------------------------------------------------------
__global__ void proj_relu_kernel(const float* __restrict__ x,
                                 const float* __restrict__ Win,
                                 const float* __restrict__ bin,
                                 float* __restrict__ h0, int N) {
    int tid = blockIdx.x * blockDim.x + threadIdx.x;
    if (tid >= N * 64) return;
    int n = tid >> 6, j = tid & 63;
    float s = bin[j];
#pragma unroll
    for (int f = 0; f < 6; ++f) s += x[n * 6 + f] * Win[f * 64 + j];
    h0[tid] = fmaxf(s, 0.0f);
}

// ---------------------------------------------------------------------------
// 2) Pre-swizzle concatenated weights [Wl;Wr] (128 x M) into WMMA B-fragment
//    order: Wswz[((tile*4 + kc)*32 + lane)*16 + i] = W[kc*32 + (lane>>4)*16+i]
//    [tile*16 + (lane&15)].  Each lane's 16 halves are contiguous (32 B), so
//    the GEMM fetches a whole B fragment with 2x global_load_b128.
// ---------------------------------------------------------------------------
__global__ void prep_wswz_kernel(const float* __restrict__ Wl,
                                 const float* __restrict__ Wr,
                                 _Float16* __restrict__ Wswz, int M) {
    int tid = blockIdx.x * blockDim.x + threadIdx.x;
    int total = M * 128;                       // (M/16) tiles * 4 kc * 32 * 16
    if (tid >= total) return;
    int i    = tid & 15;
    int lane = (tid >> 4) & 31;
    int kc   = (tid >> 9) & 3;
    int tile = tid >> 11;
    int k = kc * 32 + (lane >> 4) * 16 + i;
    int m = tile * 16 + (lane & 15);
    float v = (k < 64) ? Wl[k * M + m] : Wr[(k - 64) * M + m];
    Wswz[tid] = (_Float16)v;
}

// ---------------------------------------------------------------------------
// 3) Degree: deg[dst] += 1   (computed once; graph identical both layers)
// ---------------------------------------------------------------------------
__global__ void deg_kernel(const long long* __restrict__ dst,
                           float* __restrict__ deg, int E) {
    int e = blockIdx.x * blockDim.x + threadIdx.x;
    if (e < E) atomicAdd(&deg[(int)dst[e]], 1.0f);
}

// ---------------------------------------------------------------------------
// 4) Scatter-add: agg[dst] += h[src] ; 16 lanes/edge, float4 per lane.
//    h & agg (12.8 MB each) stay L2-resident on MI455X (192 MB L2).
// ---------------------------------------------------------------------------
__global__ void scatter_kernel(const long long* __restrict__ src,
                               const long long* __restrict__ dst,
                               const float* __restrict__ h,
                               float* __restrict__ agg, int E) {
    long long tid = (long long)blockIdx.x * blockDim.x + threadIdx.x;
    int e = (int)(tid >> 4);
    if (e >= E) return;
    int q = (int)(tid & 15);
    int s = (int)src[e], d = (int)dst[e];
    const float4 v = *(const float4*)(h + (size_t)s * 64 + q * 4);
    float* p = agg + (size_t)d * 64 + q * 4;
    atomicAdd(p + 0, v.x);
    atomicAdd(p + 1, v.y);
    atomicAdd(p + 2, v.z);
    atomicAdd(p + 3, v.w);
}

// ---------------------------------------------------------------------------
// 5) Ahat = [agg * 1/max(deg,1) | h] as f16, (N x 128) row-major.
//    One pass, float4 in / v4h (8 B) out.
// ---------------------------------------------------------------------------
__global__ void build_ahat_kernel(const float* __restrict__ agg,
                                  const float* __restrict__ h,
                                  const float* __restrict__ deg,
                                  _Float16* __restrict__ Ahat, int N) {
    int tid = blockIdx.x * blockDim.x + threadIdx.x;
    if (tid >= N * 32) return;
    int n = tid >> 5, q = tid & 31;
    float4 v;
    if (q < 16) {
        v = *(const float4*)(agg + (size_t)n * 64 + q * 4);
        float inv = 1.0f / fmaxf(deg[n], 1.0f);
        v.x *= inv; v.y *= inv; v.z *= inv; v.w *= inv;
    } else {
        v = *(const float4*)(h + (size_t)n * 64 + (q - 16) * 4);
    }
    v4h o = { (_Float16)v.x, (_Float16)v.y, (_Float16)v.z, (_Float16)v.w };
    *(v4h*)(Ahat + (size_t)n * 128 + q * 4) = o;
}

// ---------------------------------------------------------------------------
// 6) SAGE GEMM via WMMA, CDNA5 async staging:
//    - 16-row A panel (16 x 128 f16, 4 KB) pulled into LDS with
//      global_load_async_to_lds_b128 (ASYNCcnt), 272 B padded row stride.
//    - A fragment: 2x ds_load_b128; B fragment: 2x global_load_b128 from the
//      pre-swizzled weights; K=128 -> 4x v_wmma_f32_16x16x32_f16 per wave.
//    One block per 16-row tile (N=50000 divisible by 16); wave w owns output
//    columns [16w, 16w+16).  All lanes active around every WMMA/async op.
// ---------------------------------------------------------------------------
template <int M>
__global__ void __launch_bounds__((M / 16) * 32)
sage_gemm_kernel(const _Float16* __restrict__ Ahat,   // (N x 128) f16
                 const _Float16* __restrict__ Wswz,   // fragment-ordered
                 const float* __restrict__ bias,      // (M)
                 float* __restrict__ out) {
    constexpr int AH = 136;                  // halves per LDS row (272 B)
    extern __shared__ _Float16 sA[];         // 16 x AH halves

    const int tid  = threadIdx.x;
    const int nthr = blockDim.x;
    const int r0   = blockIdx.x * 16;

    // ---- async stage A panel: 256 x 16-byte lane transfers ----
    for (int i = tid; i < 256; i += nthr) {  // nthr divides 256: EXEC all 1s
        int row = i >> 4, seg = i & 15;
        const _Float16* gp = Ahat + (size_t)(r0 + row) * 128 + seg * 8;
        unsigned lds_off = (unsigned)(uintptr_t)(sA + row * AH + seg * 8);
        unsigned long long ga = (unsigned long long)(uintptr_t)gp;
        asm volatile("global_load_async_to_lds_b128 %0, %1, off"
                     :: "v"(lds_off), "v"(ga) : "memory");
    }
    asm volatile("s_wait_asynccnt 0" ::: "memory");
    __syncthreads();

    const int wave = tid >> 5;
    const int lane = tid & 31;
    const int l15  = lane & 31 & 15;
    const int hsel = lane >> 4;              // 0: lanes 0-15, 1: lanes 16-31

    const _Float16* arow  = sA + l15 * AH;
    const _Float16* wbase = Wswz + ((size_t)(wave * 4) * 32 + lane) * 16;

    v8f acc = {};
#pragma unroll
    for (int kc = 0; kc < 4; ++kc) {
        // A fragment: halves 0-15 = K kbase+hsel*16..+15 (matches B swizzle).
        // 16-byte aligned ds_load_b128 pairs.
        const int ak = kc * 32 + hsel * 16;
        v8h a0 = *(const v8h*)(arow + ak);
        v8h a1 = *(const v8h*)(arow + ak + 8);
        v16h a = __builtin_shufflevector(a0, a1, 0, 1, 2, 3, 4, 5, 6, 7,
                                         8, 9, 10, 11, 12, 13, 14, 15);
        // B fragment: lane's 16 halves contiguous in pre-swizzled weights.
        const v8h* bp = (const v8h*)(wbase + (size_t)kc * 32 * 16);
        v8h b0 = bp[0];
        v8h b1 = bp[1];
        v16h b = __builtin_shufflevector(b0, b1, 0, 1, 2, 3, 4, 5, 6, 7,
                                         8, 9, 10, 11, 12, 13, 14, 15);

        acc = __builtin_amdgcn_wmma_f32_16x16x32_f16(
            /*neg_a=*/false, a, /*neg_b=*/false, b,
            /*c_mod=*/(short)0, acc, /*reuse_a=*/false, /*reuse_b=*/false);
    }

    // D layout: lanes 0-15 VGPR r -> row r, lanes 16-31 VGPR r -> row 8+r.
    const int ocol = wave * 16 + l15;
    const float bv = bias[ocol];
#pragma unroll
    for (int r = 0; r < 8; ++r) {
        const int orow = r0 + hsel * 8 + r;
        out[orow * M + ocol] = acc[r] + bv;
    }
}

// ---------------------------------------------------------------------------
// 7) In-place LayerNorm + ReLU, one wave32 per row, shfl reductions.
// ---------------------------------------------------------------------------
template <int D>
__global__ void ln_relu_kernel(float* __restrict__ buf,
                               const float* __restrict__ g,
                               const float* __restrict__ beta, int N) {
    const int row = blockIdx.x * (blockDim.x >> 5) + (threadIdx.x >> 5);
    if (row >= N) return;                    // wave-uniform branch
    const int lane = threadIdx.x & 31;
    constexpr int PER = D / 32;
    float v[PER];
    float s = 0.0f;
#pragma unroll
    for (int i = 0; i < PER; ++i) {
        v[i] = buf[row * D + lane + i * 32];
        s += v[i];
    }
#pragma unroll
    for (int o = 16; o >= 1; o >>= 1) s += __shfl_xor(s, o, 32);
    const float mu = s * (1.0f / D);
    float var = 0.0f;
#pragma unroll
    for (int i = 0; i < PER; ++i) {
        float d = v[i] - mu;
        var += d * d;
    }
#pragma unroll
    for (int o = 16; o >= 1; o >>= 1) var += __shfl_xor(var, o, 32);
    var *= (1.0f / D);
    const float rstd = rsqrtf(var + LN_EPS);
#pragma unroll
    for (int i = 0; i < PER; ++i) {
        const int c = lane + i * 32;
        float y = (v[i] - mu) * rstd * g[c] + beta[c];
        buf[row * D + c] = fmaxf(y, 0.0f);
    }
}

// ---------------------------------------------------------------------------
extern "C" void kernel_launch(void* const* d_in, const int* in_sizes, int n_in,
                              void* d_out, int out_size, void* d_ws, size_t ws_size,
                              hipStream_t stream) {
    const float*     x    = (const float*)d_in[0];
    const long long* ei   = (const long long*)d_in[1];   // (2, NE) int64
    const float*     Win  = (const float*)d_in[2];
    const float*     bin  = (const float*)d_in[3];
    const float*     Wl1  = (const float*)d_in[4];
    const float*     bl1  = (const float*)d_in[5];
    const float*     Wr1  = (const float*)d_in[6];
    const float*     g1   = (const float*)d_in[7];
    const float*     be1  = (const float*)d_in[8];
    const float*     Wl2  = (const float*)d_in[9];
    const float*     bl2  = (const float*)d_in[10];
    const float*     Wr2  = (const float*)d_in[11];
    const float*     g2   = (const float*)d_in[12];
    const float*     be2  = (const float*)d_in[13];

    const int N  = in_sizes[0] / 6;      // 50000
    const int NE = in_sizes[1] / 2;      // 800000
    const long long* src = ei;
    const long long* dst = ei + NE;

    // Workspace layout (~51.5 MB total)
    const size_t szH = (size_t)N * 64 * sizeof(float);   // 12.8 MB
    char* ws = (char*)d_ws;
    float*    h0   = (float*)(ws);
    float*    h1   = (float*)(ws + szH);
    float*    agg  = (float*)(ws + 2 * szH);
    _Float16* Ahat = (_Float16*)(ws + 3 * szH);          // N*128 f16 == szH
    float*    deg  = (float*)(ws + 4 * szH);
    size_t degPad  = ((size_t)N * sizeof(float) + 255) & ~(size_t)255;
    _Float16* Ws1  = (_Float16*)(ws + 4 * szH + degPad); // 64*128 halves
    _Float16* Ws2  = Ws1 + 64 * 128;                     // 128*128 halves

    float* out = (float*)d_out;          // (N, 128) f32

    const int BT = 256;
    const size_t smem = 16 * 136 * sizeof(_Float16);     // 4352 B

    // ---- setup ----
    hipMemsetAsync(deg, 0, (size_t)N * sizeof(float), stream);
    hipMemsetAsync(agg, 0, szH, stream);
    prep_wswz_kernel<<<(64 * 128 + BT - 1) / BT, BT, 0, stream>>>(Wl1, Wr1, Ws1, 64);
    prep_wswz_kernel<<<(128 * 128 + BT - 1) / BT, BT, 0, stream>>>(Wl2, Wr2, Ws2, 128);
    proj_relu_kernel<<<((size_t)N * 64 + BT - 1) / BT, BT, 0, stream>>>(
        x, Win, bin, h0, N);
    deg_kernel<<<(NE + BT - 1) / BT, BT, 0, stream>>>(dst, deg, NE);

    // ---- layer 1 ----
    scatter_kernel<<<(int)(((long long)NE * 16 + BT - 1) / BT), BT, 0, stream>>>(
        src, dst, h0, agg, NE);
    build_ahat_kernel<<<((size_t)N * 32 + BT - 1) / BT, BT, 0, stream>>>(
        agg, h0, deg, Ahat, N);
    sage_gemm_kernel<64><<<N / 16, (64 / 16) * 32, smem, stream>>>(
        Ahat, Ws1, bl1, h1);
    ln_relu_kernel<64><<<(N + 7) / 8, 256, 0, stream>>>(h1, g1, be1, N);

    // ---- layer 2 ----
    hipMemsetAsync(agg, 0, szH, stream);
    scatter_kernel<<<(int)(((long long)NE * 16 + BT - 1) / BT), BT, 0, stream>>>(
        src, dst, h1, agg, NE);
    build_ahat_kernel<<<((size_t)N * 32 + BT - 1) / BT, BT, 0, stream>>>(
        agg, h1, deg, Ahat, N);
    sage_gemm_kernel<128><<<N / 16, (128 / 16) * 32, smem, stream>>>(
        Ahat, Ws2, bl2, out);
    ln_relu_kernel<128><<<(N + 7) / 8, 256, 0, stream>>>(out, g2, be2, N);
}
